// YoloPostProcess_73126113181872
// MI455X (gfx1250) — matrix operand order repaired and checked by reading, not compiled, not run on hardware
//
#include <hip/hip_runtime.h>
#include <hip/hip_bf16.h>

typedef __attribute__((ext_vector_type(16))) _Float16 v16h;
typedef __attribute__((ext_vector_type(8)))  float    v8f;

#define NCQ 6
#define REG_MAXQ 16
#define ATOT 6300
#define A0N 4800
#define A1N 1200
#define A2N 300
#define BQ 64
#define MAX_DETQ 300
#define CONF_T 0.1f
#define IOU_T 0.5f
#define MAX_WHQ 7680.0f
#define NOQ 70
#define SORT_N 8192

__device__ __forceinline__ void anchor_info(int a, int b,
    const float* __restrict__ x0, const float* __restrict__ x1, const float* __restrict__ x2,
    const float** base, int* cs, float* ax, float* ay, float* st)
{
  if (a < A0N) {
    *base = x0 + (size_t)b * NOQ * A0N + a;
    *cs = A0N;
    int gy = a / 80, gx = a - gy * 80;
    *ax = gx + 0.5f; *ay = gy + 0.5f; *st = 8.0f;
  } else if (a < A0N + A1N) {
    int i = a - A0N;
    *base = x1 + (size_t)b * NOQ * A1N + i;
    *cs = A1N;
    int gy = i / 40, gx = i - gy * 40;
    *ax = gx + 0.5f; *ay = gy + 0.5f; *st = 16.0f;
  } else {
    int i = a - (A0N + A1N);
    *base = x2 + (size_t)b * NOQ * A2N + i;
    *cs = A2N;
    int gy = i / 20, gx = i - gy * 20;
    *ax = gx + 0.5f; *ay = gy + 0.5f; *st = 32.0f;
  }
}

// ---------------------------------------------------------------------------
// Kernel 1: DFL decode via WMMA + sigmoid class max. One wave = 16 anchors.
// A (16x32 f16): rows = anchors, K 0..15 = exp(logits) of coord c, K 16..31 =
// exp(logits) of coord c+1 (per the ISA 16-bit A layout: lane m holds bins
// 0..7 of row m, lane m+16 holds bins 8..15).
// B (32x16 f16) constant selector: col0 = 1 (k<16), col1 = k (k<16),
// col2 = 1 (k>=16), col3 = k-16 (k>=16)  ->  D[m,0..3] = (Σe, Σe·r) x 2 coords.
// ---------------------------------------------------------------------------
__global__ __launch_bounds__(256) void yolo_decode_wmma(
    const float* __restrict__ x0, const float* __restrict__ x1, const float* __restrict__ x2,
    float* __restrict__ wbox, float* __restrict__ wsc, float* __restrict__ wlb)
{
  __shared__ float ldsD[8][16][8];
  const int tid  = threadIdx.x;
  const int w    = tid >> 5;
  const int lane = tid & 31;
  const int m    = lane & 15;      // anchor-in-group / B column / D column
  const int hi   = lane >> 4;      // which K half this lane supplies
  const int b    = blockIdx.y;
  const int a0   = blockIdx.x * 128 + w * 16;
  const int a    = a0 + m;
  const int aa   = a < ATOT ? a : ATOT - 1;

  const float* base; int cs; float ax, ay, st;
  anchor_info(aa, b, x0, x1, x2, &base, &cs, &ax, &ay, &st);

  // prefetch the class logits for this anchor (global_prefetch_b8)
  __builtin_prefetch(base + (size_t)(4 * REG_MAXQ) * cs, 0, 0);

  // constant B fragment (identical for both WMMAs)
  v16h bf;
#pragma unroll
  for (int e = 0; e < 16; ++e) {
    float bv = 0.0f;
    if (!(e >> 3)) { if (m == 0) bv = 1.0f; else if (m == 1) bv = (float)(hi * 8 + (e & 7)); }
    else           { if (m == 2) bv = 1.0f; else if (m == 3) bv = (float)(hi * 8 + (e & 7)); }
    bf[e] = (_Float16)bv;
  }

  // ---- WMMA #1: coords 0 (l) and 1 (t) ----
  v16h af;
#pragma unroll
  for (int e = 0; e < 8; ++e) {
    int r = hi * 8 + e;
    af[e]     = (_Float16)__expf(base[(size_t)(0 * REG_MAXQ + r) * cs]);
    af[8 + e] = (_Float16)__expf(base[(size_t)(1 * REG_MAXQ + r) * cs]);
  }
  v8f cz = {};
  v8f d1 = __builtin_amdgcn_wmma_f32_16x16x32_f16(false, af, false, bf,
                                                  (short)0, cz, false, false);

  // ---- WMMA #2: coords 2 (r) and 3 (b) ----
#pragma unroll
  for (int e = 0; e < 8; ++e) {
    int r = hi * 8 + e;
    af[e]     = (_Float16)__expf(base[(size_t)(2 * REG_MAXQ + r) * cs]);
    af[8 + e] = (_Float16)__expf(base[(size_t)(3 * REG_MAXQ + r) * cs]);
  }
  v8f d2 = __builtin_amdgcn_wmma_f32_16x16x32_f16(false, af, false, bf,
                                                  (short)0, cz, false, false);

  // D layout: VGPR j -> row j (lanes 0-15) / row 8+j (lanes 16-31), col = m.
  if (m < 4) {
#pragma unroll
    for (int j = 0; j < 8; ++j) {
      int row = hi ? 8 + j : j;
      ldsD[w][row][m]     = d1[j];   // cols 0..3: Σe_l, Σe·r_l, Σe_t, Σe·r_t
      ldsD[w][row][4 + m] = d2[j];   // cols 4..7: Σe_r, Σe·r_r, Σe_b, Σe·r_b
    }
  }
  __syncthreads();

  if (hi == 0 && a < ATOT) {
    float dl = ldsD[w][m][1] / ldsD[w][m][0];
    float dt = ldsD[w][m][3] / ldsD[w][m][2];
    float dr = ldsD[w][m][5] / ldsD[w][m][4];
    float db = ldsD[w][m][7] / ldsD[w][m][6];

    // xyxy * stride (what the NMS stage consumes)
    float bx1 = (ax - dl) * st;
    float by1 = (ay - dt) * st;
    float bx2 = (ax + dr) * st;
    float by2 = (ay + db) * st;

    // class max + argmax (sigmoid is monotonic)
    float mz = base[(size_t)(4 * REG_MAXQ) * cs];
    int   ml = 0;
#pragma unroll
    for (int j = 1; j < NCQ; ++j) {
      float z = base[(size_t)(4 * REG_MAXQ + j) * cs];
      if (z > mz) { mz = z; ml = j; }
    }
    float score = 1.0f / (1.0f + __expf(-mz));

    size_t o = (size_t)b * ATOT + a;
    wbox[o * 4 + 0] = bx1;
    wbox[o * 4 + 1] = by1;
    wbox[o * 4 + 2] = bx2;
    wbox[o * 4 + 3] = by2;
    wsc[o] = score;
    wlb[o] = (float)ml;
  }
}

// ---------------------------------------------------------------------------
// Kernel 2: per-image bitonic top-k sort (exact lax.top_k order) + greedy NMS
// + rescale/clip + output packing. One 1024-thread block per image; 64KB LDS.
// ---------------------------------------------------------------------------
__global__ __launch_bounds__(1024) void yolo_sort_nms(
    const float* __restrict__ wbox, const float* __restrict__ wsc,
    const float* __restrict__ wlb, const int* __restrict__ tsz,
    float* __restrict__ out)
{
  __shared__ unsigned long long keys[SORT_N];   // 64 KB, reused for NMS arrays
  const int tid = threadIdx.x;
  const int b   = blockIdx.x;

  // pack (masked score desc, index asc) into one monotone u64 key
#pragma unroll
  for (int rep = 0; rep < SORT_N / 1024; ++rep) {
    int e = tid + rep * 1024;
    unsigned long long k = 0ull;
    if (e < ATOT) {
      float s  = wsc[(size_t)b * ATOT + e];
      float ms = (s > CONF_T) ? s : -1.0f;
      unsigned u = __float_as_uint(ms);
      u = (u & 0x80000000u) ? ~u : (u | 0x80000000u);
      k = ((unsigned long long)u << 32) |
          (unsigned long long)(0xFFFFFFFFu - (unsigned)e);
    }
    keys[e] = k;
  }
  __syncthreads();

  // bitonic sort, descending
  for (int k2 = 2; k2 <= SORT_N; k2 <<= 1) {
    for (int jj = k2 >> 1; jj > 0; jj >>= 1) {
#pragma unroll
      for (int rep = 0; rep < SORT_N / 1024; ++rep) {
        int i   = tid + rep * 1024;
        int ixj = i ^ jj;
        if (ixj > i) {
          unsigned long long va = keys[i], vb = keys[ixj];
          bool descBlock = ((i & k2) == 0);
          if ((va < vb) == descBlock) { keys[i] = vb; keys[ixj] = va; }
        }
      }
      __syncthreads();
    }
  }

  // extract top-300, gather candidate data
  unsigned long long mykey = 0ull;
  if (tid < MAX_DETQ) mykey = keys[tid];
  __syncthreads();

  float px1 = 0, py1 = 0, px2 = 0, py2 = 0, sc = -1.0f, lb = 0;
  if (tid < MAX_DETQ) {
    int idx = (int)(0xFFFFFFFFu - (unsigned)(mykey & 0xFFFFFFFFull));
    idx = idx < 0 ? 0 : (idx >= ATOT ? ATOT - 1 : idx);
    size_t o = (size_t)b * ATOT + idx;
    px1 = wbox[o * 4 + 0]; py1 = wbox[o * 4 + 1];
    px2 = wbox[o * 4 + 2]; py2 = wbox[o * 4 + 3];
    sc  = wsc[o];
    lb  = wlb[o];
  }

  // reuse the LDS buffer for the NMS working set
  float* F  = (float*)keys;
  float* X1 = F;           float* Y1 = F + 512;
  float* X2 = F + 1024;    float* Y2 = F + 1536;
  float* AR = F + 2048;
  int*   KP = (int*)(F + 2560);

  if (tid < MAX_DETQ) {
    float off = lb * MAX_WHQ;                     // class-offset trick
    float ox1 = px1 + off, oy1 = py1 + off;
    float ox2 = px2 + off, oy2 = py2 + off;
    X1[tid] = ox1; Y1[tid] = oy1; X2[tid] = ox2; Y2[tid] = oy2;
    AR[tid] = (ox2 - ox1) * (oy2 - oy1);
    KP[tid] = (sc > CONF_T) ? 1 : 0;
  }
  __syncthreads();

  // greedy NMS, 300 sequential rounds, parallel suppression per round
  for (int i = 0; i < MAX_DETQ; ++i) {
    if (tid < MAX_DETQ && tid > i && KP[i]) {
      float ix1 = fmaxf(X1[i], X1[tid]);
      float iy1 = fmaxf(Y1[i], Y1[tid]);
      float ix2 = fminf(X2[i], X2[tid]);
      float iy2 = fminf(Y2[i], Y2[tid]);
      float inter = fmaxf(ix2 - ix1, 0.0f) * fmaxf(iy2 - iy1, 0.0f);
      float iou = inter / (AR[i] + AR[tid] - inter + 1e-7f);
      if (iou > IOU_T) KP[tid] = 0;
    }
    __syncthreads();
  }

  // rescale to target size, clip, pack outputs
  if (tid < MAX_DETQ) {
    float th = (float)tsz[b * 2 + 0];
    float tw = (float)tsz[b * 2 + 1];
    float gain = fminf(480.0f / th, 640.0f / tw);
    float padx = (640.0f - tw * gain) * 0.5f;
    float pady = (480.0f - th * gain) * 0.5f;

    float cx1 = fminf(fmaxf((px1 - padx) / gain, 0.0f), tw);
    float cy1 = fminf(fmaxf((py1 - pady) / gain, 0.0f), th);
    float cx2 = fminf(fmaxf((px2 - padx) / gain, 0.0f), tw);
    float cy2 = fminf(fmaxf((py2 - pady) / gain, 0.0f), th);

    int keep = KP[tid];
    size_t ob = (size_t)b * MAX_DETQ + tid;
    out[ob * 4 + 0] = cx1;
    out[ob * 4 + 1] = cy1;
    out[ob * 4 + 2] = cx2;
    out[ob * 4 + 3] = cy2;
    out[(size_t)BQ * MAX_DETQ * 4 + ob] = keep ? sc : 0.0f;   // scores
    out[(size_t)BQ * MAX_DETQ * 5 + ob] = keep ? lb : -1.0f;  // labels
  }
}

extern "C" void kernel_launch(void* const* d_in, const int* in_sizes, int n_in,
                              void* d_out, int out_size, void* d_ws, size_t ws_size,
                              hipStream_t stream) {
  (void)in_sizes; (void)n_in; (void)out_size; (void)ws_size;
  const float* x0  = (const float*)d_in[0];
  const float* x1  = (const float*)d_in[1];
  const float* x2  = (const float*)d_in[2];
  const int*   tsz = (const int*)d_in[3];

  float* ws   = (float*)d_ws;
  float* wbox = ws;                              // B*ATOT*4 floats
  float* wsc  = ws + (size_t)BQ * ATOT * 4;      // B*ATOT floats
  float* wlb  = ws + (size_t)BQ * ATOT * 5;      // B*ATOT floats

  dim3 g1((ATOT + 127) / 128, BQ);
  yolo_decode_wmma<<<g1, 256, 0, stream>>>(x0, x1, x2, wbox, wsc, wlb);
  yolo_sort_nms<<<BQ, 1024, 0, stream>>>(wbox, wsc, wlb, tsz, (float*)d_out);
}
